// GCBlock8_23545010717065
// MI455X (gfx1250) — matrix operand order repaired and checked by reading, not compiled, not run on hardware
//
#include <hip/hip_runtime.h>

typedef __attribute__((ext_vector_type(16))) _Float16 v16h;
typedef __attribute__((ext_vector_type(8)))  float    v8f;

__device__ __forceinline__ v8f wmma_f16(v16h a, v16h b, v8f c) {
  return __builtin_amdgcn_wmma_f32_16x16x32_f16(false, a, false, b, (short)0, c,
                                                false, false);
}

// ---------------------------------------------------------------------------
// Stage 1+3 (tiny GEMMs on WMMA):
//   conv1d(512x18) = query^T(512x100) * Wl^T(100x18) + bl
//   qout(100x512)  = transpose( conv1d(512x18) * Wd^T(18x100) + bd )
// f32 operands split into hi/lo f16 pairs; 3-product compensated WMMA.
// One workgroup, 8 waves, each wave owns 4 M-tiles of 16 rows.
// ---------------------------------------------------------------------------
__global__ __launch_bounds__(256) void gemm_stage(
    const float* __restrict__ query,   // (100, 512) : q major, bc minor
    const float* __restrict__ Wl,      // (18, 100)
    const float* __restrict__ bl,      // (18)
    const float* __restrict__ Wd,      // (100, 18)
    const float* __restrict__ bd,      // (100)
    float* __restrict__ conv1d_ws,     // (512, 18) workspace
    float* __restrict__ qout)          // (100, 512) = d_out[0:51200]
{
  __shared__ float sconv[512 * 18];
  const int lane = threadIdx.x & 31;
  const int wave = threadIdx.x >> 5;
  const int ml   = lane & 15;   // A: row M / B,C,D: col N
  const int kh   = lane >> 4;   // K-half selector per ISA layouts

  // ---------------- GEMM1 ----------------
  for (int mt = wave; mt < 32; mt += 8) {
    const int bc = mt * 16 + ml;
    v16h ah[4], al[4];
#pragma unroll
    for (int ks = 0; ks < 4; ++ks) {
#pragma unroll
      for (int i = 0; i < 8; ++i) {
#pragma unroll
        for (int j = 0; j < 2; ++j) {
          // 16-bit A 16x32 layout: VGPR i, lanes(kh): K = base + (i&3)*2 + j
          int k = ks * 32 + ((i < 4) ? 0 : 16) + kh * 8 + (i & 3) * 2 + j;
          float x = (k < 100) ? query[k * 512 + bc] : 0.0f;
          _Float16 h = (_Float16)x;
          ah[ks][2 * i + j] = h;
          al[ks][2 * i + j] = (_Float16)(x - (float)h);
        }
      }
    }
#pragma unroll
    for (int nt = 0; nt < 2; ++nt) {
      const int n = nt * 16 + ml;
      v8f acc = {};
#pragma unroll
      for (int ks = 0; ks < 4; ++ks) {
        v16h bh, blo;
#pragma unroll
        for (int h = 0; h < 16; ++h) {
          // 16-bit B 32x16 layout: half h, lane-half kh: K = (h<8?0:16) + kh*8 + (h&7)
          int k = ks * 32 + ((h < 8) ? 0 : 16) + kh * 8 + (h & 7);
          float x = (k < 100 && n < 18) ? Wl[n * 100 + k] : 0.0f;
          _Float16 hh = (_Float16)x;
          bh[h]  = hh;
          blo[h] = (_Float16)(x - (float)hh);
        }
        acc = wmma_f16(ah[ks], bh,  acc);
        acc = wmma_f16(ah[ks], blo, acc);
        acc = wmma_f16(al[ks], bh,  acc);
      }
      if (n < 18) {
        float bln = bl[n];
#pragma unroll
        for (int r = 0; r < 8; ++r) {
          int gbc = mt * 16 + r + 8 * kh;   // D layout: M = r + 8*kh
          float v = acc[r] + bln;
          sconv[gbc * 18 + n]     = v;
          conv1d_ws[gbc * 18 + n] = v;
        }
      }
    }
  }
  __syncthreads();

  // ---------------- GEMM2 ----------------
  for (int mt = wave; mt < 32; mt += 8) {
    const int bc = mt * 16 + ml;
    v16h a2h, a2l;
#pragma unroll
    for (int i = 0; i < 8; ++i) {
#pragma unroll
      for (int j = 0; j < 2; ++j) {
        int k = ((i < 4) ? 0 : 16) + kh * 8 + (i & 3) * 2 + j;  // K padded 18->32
        float x = (k < 18) ? sconv[bc * 18 + k] : 0.0f;
        _Float16 h = (_Float16)x;
        a2h[2 * i + j] = h;
        a2l[2 * i + j] = (_Float16)(x - (float)h);
      }
    }
    for (int nt = 0; nt < 7; ++nt) {        // N padded 100 -> 112
      const int n = nt * 16 + ml;
      v16h bh, blo;
#pragma unroll
      for (int h = 0; h < 16; ++h) {
        int k = ((h < 8) ? 0 : 16) + kh * 8 + (h & 7);
        float x = (k < 18 && n < 100) ? Wd[n * 18 + k] : 0.0f;
        _Float16 hh = (_Float16)x;
        bh[h]  = hh;
        blo[h] = (_Float16)(x - (float)hh);
      }
      v8f acc = {};
      acc = wmma_f16(a2h, bh,  acc);
      acc = wmma_f16(a2h, blo, acc);
      acc = wmma_f16(a2l, bh,  acc);
      if (n < 100) {
        float bdn = bd[n];
#pragma unroll
        for (int r = 0; r < 8; ++r) {
          int gbc = mt * 16 + r + 8 * kh;
          qout[n * 512 + gbc] = acc[r] + bdn;   // (100,B,C) transposed store
        }
      }
    }
  }
}

// ---------------------------------------------------------------------------
// Stage 2 (memory-bound stencil): one 64x64 tile of one (b,c) plane per block.
// value read once (halo-2 tile in LDS), y written once. ~11.5us HBM floor.
// ---------------------------------------------------------------------------
__global__ __launch_bounds__(256) void dwconv_stage(
    const float* __restrict__ value,   // (512, 256, 256)
    const float* __restrict__ conv1d,  // (512, 18)
    float* __restrict__ y)             // (512, 256, 256)
{
  const int plane = blockIdx.y;
  const int tx = (blockIdx.x & 3) * 64;
  const int ty = (blockIdx.x >> 2) * 64;
  const float* vp = value + (size_t)plane * 65536u;
  float*       yp = y     + (size_t)plane * 65536u;

  __shared__ float sval[68][72];   // value, halo 2, origin (-2,-2)
  __shared__ float sdiff[66][72];  // exp(-(v-y_temp)^2), halo 1, origin (-1,-1)
  __shared__ float svd[66][72];    // value*diff, halo 1
  __shared__ float sw[18];

  if (threadIdx.x < 18) sw[threadIdx.x] = conv1d[plane * 18 + threadIdx.x];

  for (int i = threadIdx.x; i < 68 * 68; i += 256) {
    int r = i / 68, c = i - r * 68;
    int gh = ty + r - 2, gw = tx + c - 2;
    float v = 0.0f;
    if ((unsigned)gh < 256u && (unsigned)gw < 256u) v = vp[gh * 256 + gw];
    sval[r][c] = v;
  }
  __syncthreads();

  float fm[9], wk[9], f2[9], wd[3];
#pragma unroll
  for (int i = 0; i < 9; ++i) {
    fm[i] = sw[i];
    wk[i] = sw[9 + i];
    f2[i] = fabsf(wk[i]);
  }
  wd[0] = wk[0] + wk[3] + wk[6];
  wd[1] = wk[1] + wk[4] + wk[7];
  wd[2] = wk[2] + wk[5] + wk[8];

  // Phase 1: diff & v*diff on the 66x66 extended region. Out-of-image
  // entries are 0 (SAME conv zero-pads the diff/vd arrays themselves).
  for (int i = threadIdx.x; i < 66 * 66; i += 256) {
    int r = i / 66, c = i - r * 66;
    int gh = ty + r - 1, gw = tx + c - 1;
    float dv = 0.0f, vdv = 0.0f;
    if ((unsigned)gh < 256u && (unsigned)gw < 256u) {
      float yt = 0.0f;
#pragma unroll
      for (int ki = 0; ki < 3; ++ki)
#pragma unroll
        for (int kj = 0; kj < 3; ++kj)
          yt = fmaf(fm[ki * 3 + kj], sval[r + ki][c + kj], yt);
      float v = sval[r + 1][c + 1];
      float d = v - yt;
      dv  = __expf(-d * d);
      vdv = v * dv;
    }
    sdiff[r][c] = dv;
    svd[r][c]   = vdv;
  }
  __syncthreads();

  // Phase 2: 16 output pixels per thread.
  const int cx  = threadIdx.x & 63;
  const int ry0 = threadIdx.x >> 6;          // 0..3
  for (int rr = 0; rr < 16; ++rr) {
    int ryy = ry0 * 16 + rr;
    float yt = 0.0f, y9 = 0.0f, res = 0.0f;
#pragma unroll
    for (int ki = 0; ki < 3; ++ki)
#pragma unroll
      for (int kj = 0; kj < 3; ++kj) {
        yt  = fmaf(fm[ki * 3 + kj], sval[ryy + ki + 1][cx + kj + 1], yt);
        y9  = fmaf(f2[ki * 3 + kj], sdiff[ryy + ki][cx + kj], y9);
        res = fmaf(wk[ki * 3 + kj], svd[ryy + ki][cx + kj], res);
      }
    float resd = wd[0] * svd[ryy + 1][cx]
               + wd[1] * svd[ryy + 1][cx + 1]
               + wd[2] * svd[ryy + 1][cx + 2];
    float out = yt - (res - resd) / (y9 + 1e-10f);
    yp[(ty + ryy) * 256 + tx + cx] = out;
  }
}

extern "C" void kernel_launch(void* const* d_in, const int* in_sizes, int n_in,
                              void* d_out, int out_size, void* d_ws, size_t ws_size,
                              hipStream_t stream) {
  const float* query = (const float*)d_in[0];  // (100,8,64)
  const float* value = (const float*)d_in[1];  // (8,64,256,256)
  // d_in[2..4]: unused scalars
  const float* Wl = (const float*)d_in[5];     // (18,100)
  const float* bl = (const float*)d_in[6];     // (18)
  const float* Wd = (const float*)d_in[7];     // (100,18)
  const float* bd = (const float*)d_in[8];     // (100)

  float* out    = (float*)d_out;               // q (51200 f32) then y (33.5M f32)
  float* conv1d = (float*)d_ws;                // 512*18 f32 scratch

  gemm_stage<<<1, 256, 0, stream>>>(query, Wl, bl, Wd, bd, conv1d, out);
  dwconv_stage<<<dim3(16, 512), 256, 0, stream>>>(value, conv1d, out + 51200);
}
